// PeriodicAttentionLayer_RoPE_81776177316185
// MI455X (gfx1250) — matrix operand order repaired and checked by reading, not compiled, not run on hardware
//
#include <hip/hip_runtime.h>
#include <hip/hip_bf16.h>

// ---------------------------------------------------------------------------
// PeriodicAttentionLayer_RoPE for MI455X (gfx1250, wave32, WMMA)
// B=8 T=96 H=32 W=32 D=64, SEG=24, NH=8 -> R=8192 rows, n=4 segs,
// in_dim=1536, hid=768, hd=96, M = R*n = 32768.
// Pipeline: gather->bf16, W transposed->bf16, WMMA GEMM (QKV fused N=2304,
// bf16 out), per-(row,head) wave attention with RoPE, WMMA GEMM out + scatter.
// GEMM staging uses gfx1250 async global->LDS loads (double buffered) when
// the toolchain exposes the builtins; otherwise falls back to sync staging.
// ---------------------------------------------------------------------------

typedef __bf16 bf16;
typedef __attribute__((ext_vector_type(16))) __bf16 v16bf;
typedef __attribute__((ext_vector_type(8)))  float  v8f;
typedef __attribute__((ext_vector_type(4)))  float  f32x4;
typedef int v4i __attribute__((vector_size(16)));

union FragBF { uint4 u[2]; v16bf v; };
union Pack4  { bf16 h[4]; uint2 u; };

#define R_ROWS   8192
#define M_TOT    32768
#define K_IN     1536      // in_dim
#define N_QKV    2304      // 3*hid
#define HID      768
#define HD       96
#define LDSPAD   40        // 32 + 8 bf16 pad, rows stay 16B aligned (80B)
#define KTILE    32

#ifndef __has_builtin
#define __has_builtin(x) 0
#endif
#if __has_builtin(__builtin_amdgcn_global_load_async_to_lds_b128) && \
    __has_builtin(__builtin_amdgcn_s_wait_asynccnt)
#define USE_ASYNC_LDS 1
#else
#define USE_ASYNC_LDS 0
#endif

#if USE_ASYNC_LDS
__device__ __forceinline__ void async_b128(bf16* lds, const bf16* g) {
  __builtin_amdgcn_global_load_async_to_lds_b128(
      (__attribute__((address_space(1))) v4i*)g,
      (__attribute__((address_space(3))) v4i*)lds,
      0, 0);
}
#endif

// ---------------------------------------------------------------------------
// 1) Gather x (B,T,H,W,D f32) -> Xs[m][e] bf16, m = ((b*32+h)*32+w)*4+seg,
//    e = p*64+d with t = seg*24+p. Streaming read -> nontemporal.
// ---------------------------------------------------------------------------
__global__ __launch_bounds__(256) void gather_x_kernel(
    const float* __restrict__ x, bf16* __restrict__ Xs)
{
  const size_t tid = (size_t)blockIdx.x * 256 + threadIdx.x;   // one per 4 elems
  if (tid >= (size_t)M_TOT * K_IN / 4) return;
  const size_t base = tid * 4;
  const int m = (int)(base / K_IN);
  const int e = (int)(base % K_IN);
  const int seg = m & 3, rw = m >> 2;
  const int w = rw & 31, hh = (rw >> 5) & 31, b = rw >> 10;
  const int p = e >> 6, d = e & 63;
  const int tt = seg * 24 + p;
  const f32x4 val = __builtin_nontemporal_load(
      (const f32x4*)(x + ((((size_t)b * 96 + tt) * 32 + hh) * 32 + w) * 64 + d));
  Pack4 pk;
  pk.h[0] = (bf16)val.x; pk.h[1] = (bf16)val.y;
  pk.h[2] = (bf16)val.z; pk.h[3] = (bf16)val.w;
  *(uint2*)(Xs + base) = pk.u;
}

// ---------------------------------------------------------------------------
// 2) Weight converts: WqkvT[j][k] (j in 0..2303 over [Wq|Wk|Wv]), WoT[n][k].
//    Transposed so GEMM B-fragments are contiguous-K ds_load_b128s.
// ---------------------------------------------------------------------------
__global__ __launch_bounds__(256) void cvt_wqkvT_kernel(
    const float* __restrict__ Wq, const float* __restrict__ Wk,
    const float* __restrict__ Wv, bf16* __restrict__ Wt)
{
  const int tid = blockIdx.x * 256 + threadIdx.x;
  if (tid >= N_QKV * K_IN) return;
  const int j = tid / K_IN, kk = tid % K_IN;
  const float* src = (j < HID) ? Wq : ((j < 2 * HID) ? Wk : Wv);
  const int jj = j % HID;
  Wt[(size_t)j * K_IN + kk] = (bf16)src[(size_t)kk * HID + jj];
}

__global__ __launch_bounds__(256) void cvt_woT_kernel(
    const float* __restrict__ Wo, bf16* __restrict__ Wt)
{
  const int tid = blockIdx.x * 256 + threadIdx.x;
  if (tid >= K_IN * HID) return;
  const int nn = tid / HID, kk = tid % HID;
  Wt[(size_t)nn * HID + kk] = (bf16)Wo[(size_t)kk * K_IN + nn];
}

// ---------------------------------------------------------------------------
// Fragment loaders (ISA 7.12.2 layouts).
// A 16x32 bf16: lane<16 holds row=lane, K {a..a+7, 16+a..16+a+7} a=0; lane>=16 a=8.
// B 32x16 bf16: lane<16 holds col=lane, K {b..b+7, 16+b..16+b+7} b=0; lane>=16 col=lane-16, b=8.
// ---------------------------------------------------------------------------
__device__ __forceinline__ v16bf load_a_frag(const bf16* rowp, int aOff) {
  FragBF f;
  f.u[0] = *(const uint4*)(rowp + aOff);
  f.u[1] = *(const uint4*)(rowp + 16 + aOff);
  return f.v;
}
__device__ __forceinline__ v16bf load_b_frag(const bf16* colp) {
  FragBF f;
  f.u[0] = *(const uint4*)(colp);
  f.u[1] = *(const uint4*)(colp + 16);
  return f.v;
}

// ---------------------------------------------------------------------------
// Shared WMMA tile core: block tile 128(M) x 256(N), 8 waves as 2x4,
// each wave 64x64 = 4x4 WMMA tiles (16 wmma : 16 ds_load_b128 per K-step).
// ---------------------------------------------------------------------------
template <int KTOT>
__device__ __forceinline__ void wmma_tiles(
    const bf16* __restrict__ Aglb, const bf16* __restrict__ Bglb,
    int mBase, int nBase, v8f (&acc)[4][4])
{
  __shared__ bf16 As[2][128 * LDSPAD];
  __shared__ bf16 Bs[2][256 * LDSPAD];

  const int t = threadIdx.x;
  const int lane = t & 31;
  const int wv = t >> 5;
  const int mWave = (wv >> 2) * 64;
  const int nWave = (wv & 3) * 64;
  const int rowA = t >> 1;           // 128 A rows, 2 threads/row (32B each)
  const int halfA = (t & 1) * 16;
  const int rowB = t;                // 256 B rows, 1 thread/row (64B each)

  const int aOff = (lane < 16) ? 0 : 8;
  const int mA   = lane & 15;
  const int bOff = (lane < 16) ? 0 : 8;
  const int nB   = lane & 15;

  const v8f vzero = {0.f,0.f,0.f,0.f,0.f,0.f,0.f,0.f};
  #pragma unroll
  for (int i = 0; i < 4; ++i)
    #pragma unroll
    for (int j = 0; j < 4; ++j) acc[i][j] = vzero;

  const bf16* gA = Aglb + (size_t)(mBase + rowA) * KTOT + halfA;
  const bf16* gB = Bglb + (size_t)(nBase + rowB) * KTOT;
  const int NK = KTOT / KTILE;

#if USE_ASYNC_LDS
  // ------------------- async double-buffered staging ----------------------
  {
    bf16* a = &As[0][0]; bf16* b = &Bs[0][0];
    async_b128(a + rowA * LDSPAD + halfA,     gA);
    async_b128(a + rowA * LDSPAD + halfA + 8, gA + 8);
    async_b128(b + rowB * LDSPAD + 0,  gB + 0);
    async_b128(b + rowB * LDSPAD + 8,  gB + 8);
    async_b128(b + rowB * LDSPAD + 16, gB + 16);
    async_b128(b + rowB * LDSPAD + 24, gB + 24);
  }
  for (int kt = 0; kt < NK; ++kt) {
    __builtin_amdgcn_s_wait_asynccnt(0);   // my part of tile kt is in LDS
    __syncthreads();                        // everyone's part is in LDS;
                                            // buf[kt^1] no longer being read
    if (kt + 1 < NK) {
      const bf16* gA2 = gA + (size_t)(kt + 1) * KTILE;
      const bf16* gB2 = gB + (size_t)(kt + 1) * KTILE;
      bf16* a = &As[(kt + 1) & 1][0]; bf16* b = &Bs[(kt + 1) & 1][0];
      async_b128(a + rowA * LDSPAD + halfA,     gA2);
      async_b128(a + rowA * LDSPAD + halfA + 8, gA2 + 8);
      async_b128(b + rowB * LDSPAD + 0,  gB2 + 0);
      async_b128(b + rowB * LDSPAD + 8,  gB2 + 8);
      async_b128(b + rowB * LDSPAD + 16, gB2 + 16);
      async_b128(b + rowB * LDSPAD + 24, gB2 + 24);
    }
    const bf16* as = &As[kt & 1][0];
    const bf16* bs = &Bs[kt & 1][0];
    v16bf af[4], bfr[4];
    #pragma unroll
    for (int i = 0; i < 4; ++i)
      af[i] = load_a_frag(as + (mWave + i * 16 + mA) * LDSPAD, aOff);
    #pragma unroll
    for (int j = 0; j < 4; ++j)
      bfr[j] = load_b_frag(bs + (nWave + j * 16 + nB) * LDSPAD + bOff);
    #pragma unroll
    for (int i = 0; i < 4; ++i)
      #pragma unroll
      for (int j = 0; j < 4; ++j)
        acc[i][j] = __builtin_amdgcn_wmma_f32_16x16x32_bf16(
            false, af[i], false, bfr[j], (short)0, acc[i][j], false, false);
  }
#else
  // ----------------------- sync fallback staging --------------------------
  bf16* sA = &As[0][0] + rowA * LDSPAD + halfA;
  bf16* sB = &Bs[0][0] + rowB * LDSPAD;
  const bf16* pA = gA;
  const bf16* pB = gB;
  for (int kt = 0; kt < NK; ++kt) {
    uint4 a0 = *(const uint4*)(pA);
    uint4 a1 = *(const uint4*)(pA + 8);
    uint4 b0 = *(const uint4*)(pB);
    uint4 b1 = *(const uint4*)(pB + 8);
    uint4 b2 = *(const uint4*)(pB + 16);
    uint4 b3 = *(const uint4*)(pB + 24);
    *(uint4*)(sA)      = a0;  *(uint4*)(sA + 8)  = a1;
    *(uint4*)(sB)      = b0;  *(uint4*)(sB + 8)  = b1;
    *(uint4*)(sB + 16) = b2;  *(uint4*)(sB + 24) = b3;
    pA += KTILE; pB += KTILE;
    __syncthreads();
    const bf16* as = &As[0][0];
    const bf16* bs = &Bs[0][0];
    v16bf af[4], bfr[4];
    #pragma unroll
    for (int i = 0; i < 4; ++i)
      af[i] = load_a_frag(as + (mWave + i * 16 + mA) * LDSPAD, aOff);
    #pragma unroll
    for (int j = 0; j < 4; ++j)
      bfr[j] = load_b_frag(bs + (nWave + j * 16 + nB) * LDSPAD + bOff);
    #pragma unroll
    for (int i = 0; i < 4; ++i)
      #pragma unroll
      for (int j = 0; j < 4; ++j)
        acc[i][j] = __builtin_amdgcn_wmma_f32_16x16x32_bf16(
            false, af[i], false, bfr[j], (short)0, acc[i][j], false, false);
    __syncthreads();
  }
#endif
}

// ---------------------------------------------------------------------------
// 3) QKV GEMM: QKV[M_TOT x N_QKV] bf16 = Xs(bf16) x WqkvT^T(bf16).
// ---------------------------------------------------------------------------
__global__ __launch_bounds__(256) void gemm_qkv_kernel(
    const bf16* __restrict__ A, const bf16* __restrict__ Bt,
    bf16* __restrict__ C)
{
  v8f acc[4][4];
  const int mBase = blockIdx.y * 128;
  const int nBase = blockIdx.x * 256;
  wmma_tiles<K_IN>(A, Bt, mBase, nBase, acc);

  const int lane = threadIdx.x & 31;
  const int wv = threadIdx.x >> 5;
  // C layout: VGPR r, lane l -> M = r + 8*(l/16), N = l%16
  const int mc = mBase + (wv >> 2) * 64 + 8 * (lane >> 4);
  const int nc = nBase + (wv & 3) * 64 + (lane & 15);
  #pragma unroll
  for (int i = 0; i < 4; ++i)
    #pragma unroll
    for (int j = 0; j < 4; ++j) {
      bf16* cp = C + (size_t)(mc + i * 16) * N_QKV + nc + j * 16;
      #pragma unroll
      for (int r = 0; r < 8; ++r) cp[(size_t)r * N_QKV] = (bf16)acc[i][j][r];
    }
}

// ---------------------------------------------------------------------------
// 4) Attention: one wave per (row, head). RoPE pairs are lane-adjacent
//    (c = lane + 32r keeps pair parity == lane parity) -> shfl_xor(1).
// ---------------------------------------------------------------------------
__global__ __launch_bounds__(256) void attn_kernel(
    const bf16* __restrict__ QKV, const float* __restrict__ fcos,
    const float* __restrict__ fsin, bf16* __restrict__ O)
{
  const int row  = blockIdx.x;        // 0..8191
  const int t    = threadIdx.x;
  const int lane = t & 31;
  const int h    = t >> 5;            // head 0..7
  const int hb   = h * HD;

  float q[4][3], k[4][3], v[4][3], cs[4][3], sn[4][3];
  #pragma unroll
  for (int i = 0; i < 4; ++i) {
    const bf16* base = QKV + (size_t)(row * 4 + i) * N_QKV;
    #pragma unroll
    for (int r = 0; r < 3; ++r) {
      const int c = lane + 32 * r;
      q[i][r] = (float)base[hb + c];
      k[i][r] = (float)base[HID + hb + c];
      v[i][r] = (float)base[2 * HID + hb + c];
      const int jj = (hb + c) >> 1;
      cs[i][r] = fcos[i * (HID / 2) + jj];
      sn[i][r] = fsin[i * (HID / 2) + jj];
    }
  }
  const float sgn = (lane & 1) ? 1.0f : -1.0f;
  #pragma unroll
  for (int i = 0; i < 4; ++i)
    #pragma unroll
    for (int r = 0; r < 3; ++r) {
      const float qp = __shfl_xor(q[i][r], 1, 32);
      const float kp = __shfl_xor(k[i][r], 1, 32);
      q[i][r] = q[i][r] * cs[i][r] + sgn * qp * sn[i][r];
      k[i][r] = k[i][r] * cs[i][r] + sgn * kp * sn[i][r];
    }

  float sc[4][4];
  #pragma unroll
  for (int i = 0; i < 4; ++i)
    #pragma unroll
    for (int m = 0; m < 4; ++m) {
      float p = q[i][0] * k[m][0] + q[i][1] * k[m][1] + q[i][2] * k[m][2];
      #pragma unroll
      for (int off = 16; off >= 1; off >>= 1) p += __shfl_xor(p, off, 32);
      sc[i][m] = p;
    }

  const float scale = 0.1020620726159657f;  // 1/sqrt(96)
  float aw[4][4];
  #pragma unroll
  for (int i = 0; i < 4; ++i) {
    float s0 = sc[i][0] * scale, s1 = sc[i][1] * scale;
    float s2 = sc[i][2] * scale, s3 = sc[i][3] * scale;
    float mx = fmaxf(fmaxf(s0, s1), fmaxf(s2, s3));
    float e0 = __expf(s0 - mx), e1 = __expf(s1 - mx);
    float e2 = __expf(s2 - mx), e3 = __expf(s3 - mx);
    const float inv = 1.0f / (e0 + e1 + e2 + e3);
    aw[i][0] = e0 * inv; aw[i][1] = e1 * inv;
    aw[i][2] = e2 * inv; aw[i][3] = e3 * inv;
  }

  #pragma unroll
  for (int i = 0; i < 4; ++i)
    #pragma unroll
    for (int r = 0; r < 3; ++r) {
      float o = aw[i][0] * v[0][r] + aw[i][1] * v[1][r]
              + aw[i][2] * v[2][r] + aw[i][3] * v[3][r];
      O[(size_t)(row * 4 + i) * HID + hb + lane + 32 * r] = (bf16)o;
    }
}

// ---------------------------------------------------------------------------
// 5) Out GEMM: [M_TOT x 1536] = O(bf16, K=768) x WoT^T(bf16) + bo, scattered
//    directly into (B,T,H,W,D) f32 output (nontemporal streaming stores).
// ---------------------------------------------------------------------------
__global__ __launch_bounds__(256) void gemm_out_kernel(
    const bf16* __restrict__ A, const bf16* __restrict__ Bt,
    const float* __restrict__ bo, float* __restrict__ out)
{
  v8f acc[4][4];
  const int mBase = blockIdx.y * 128;
  const int nBase = blockIdx.x * 256;
  wmma_tiles<HID>(A, Bt, mBase, nBase, acc);

  const int lane = threadIdx.x & 31;
  const int wv = threadIdx.x >> 5;
  const int mc  = mBase + (wv >> 2) * 64 + 8 * (lane >> 4);
  const int nc0 = nBase + (wv & 3) * 64 + (lane & 15);
  float bov[4];
  #pragma unroll
  for (int j = 0; j < 4; ++j) bov[j] = bo[nc0 + j * 16];

  #pragma unroll
  for (int i = 0; i < 4; ++i)
    #pragma unroll
    for (int j = 0; j < 4; ++j) {
      const int e = nc0 + j * 16;         // 0..1535
      const int p = e >> 6, d = e & 63;   // t-in-seg, depth
      #pragma unroll
      for (int r = 0; r < 8; ++r) {
        const int m = mc + i * 16 + r;
        const int seg = m & 3, rw = m >> 2;
        const int w = rw & 31, hh = (rw >> 5) & 31, b = rw >> 10;
        const int tt = seg * 24 + p;
        const size_t addr = ((((size_t)b * 96 + tt) * 32 + hh) * 32 + w) * 64 + d;
        __builtin_nontemporal_store(acc[i][j][r] + bov[j], out + addr);
      }
    }
}

// ---------------------------------------------------------------------------
// Launch. Workspace layout (bytes):
//   Xs    bf16 [32768 x 1536]   @ 0          (100,663,296)
//   WqkvT bf16 [2304  x 1536]   @ 100663296  (  7,077,888)
//   WoT   bf16 [1536  x  768]   @ 107741184  (  2,359,296)
//   QKV   bf16 [32768 x 2304]   @ 110100480  (150,994,944)
//   O     bf16 [32768 x  768]   @ 261095424  ( 50,331,648)
//   total ~311.4 MB
// ---------------------------------------------------------------------------
extern "C" void kernel_launch(void* const* d_in, const int* in_sizes, int n_in,
                              void* d_out, int out_size, void* d_ws, size_t ws_size,
                              hipStream_t stream) {
  const float* x    = (const float*)d_in[0];
  const float* fcos = (const float*)d_in[1];
  const float* fsin = (const float*)d_in[2];
  const float* Wq   = (const float*)d_in[3];
  const float* Wk   = (const float*)d_in[4];
  const float* Wv   = (const float*)d_in[5];
  const float* Wo   = (const float*)d_in[6];
  const float* bo   = (const float*)d_in[7];
  float* out = (float*)d_out;

  char* ws = (char*)d_ws;
  bf16* Xs    = (bf16*)(ws + 0);
  bf16* WqkvT = (bf16*)(ws + 100663296u);
  bf16* WoT   = (bf16*)(ws + 107741184u);
  bf16* QKV   = (bf16*)(ws + 110100480u);
  bf16* O     = (bf16*)(ws + 261095424u);

  gather_x_kernel<<<49152, 256, 0, stream>>>(x, Xs);
  cvt_wqkvT_kernel<<<13824, 256, 0, stream>>>(Wq, Wk, Wv, WqkvT);
  cvt_woT_kernel<<<4608, 256, 0, stream>>>(Wo, WoT);
  gemm_qkv_kernel<<<dim3(N_QKV / 256, M_TOT / 128), 256, 0, stream>>>(Xs, WqkvT, QKV);
  attn_kernel<<<R_ROWS, 256, 0, stream>>>(QKV, fcos, fsin, O);
  gemm_out_kernel<<<dim3(K_IN / 256, M_TOT / 128), 256, 0, stream>>>(O, WoT, bo, out);
}
